// SimpleFlashAttention_55430847922688
// MI455X (gfx1250) — matrix operand order repaired and checked by reading, not compiled, not run on hardware
//
#include <hip/hip_runtime.h>
#include <hip/hip_bf16.h>

typedef __attribute__((ext_vector_type(16))) __bf16 v16bf;
typedef __attribute__((ext_vector_type(8)))  float  v8f;
typedef __attribute__((ext_vector_type(4)))  int    v4i;

#define N_EMBD   1024
#define N_HEAD   16
#define HEAD_DIM 64
#define SEQ      4096
#define BATCH    4
#define NTOK     (BATCH * SEQ)    // 16384
#define BHN      (BATCH * N_HEAD) // 64
#define ATT_SCALE 0.125f          // 64^-0.5

// ---- CDNA5 async global->LDS path (guarded; falls back to sync copies) ----
#if defined(__has_builtin)
#  if __has_builtin(__builtin_amdgcn_global_load_async_to_lds_b128) && \
      __has_builtin(__builtin_amdgcn_s_wait_asynccnt)
#    define USE_ASYNC_LDS 1
#  endif
#endif
#ifndef USE_ASYNC_LDS
#  define USE_ASYNC_LDS 0
#endif

#if USE_ASYNC_LDS
typedef __attribute__((address_space(1))) v4i* gv4i_p;   // global (prints as __device__)
typedef __attribute__((address_space(3))) v4i* lv4i_p;   // LDS
__device__ inline void async_ld16(void* lds, const void* g) {
  __builtin_amdgcn_global_load_async_to_lds_b128(
      (gv4i_p)g, (lv4i_p)lds, 0, 0);
}
#endif

__device__ inline __bf16 f2bf(float f) {
  unsigned u = __builtin_bit_cast(unsigned, f);
  unsigned r = (u + 0x7FFFu + ((u >> 16) & 1u)) >> 16;   // RNE
  unsigned short h = (unsigned short)r;
  return __builtin_bit_cast(__bf16, h);
}

__device__ inline v8f v8f_zero() {
  v8f z;
#pragma unroll
  for (int i = 0; i < 8; ++i) z[i] = 0.0f;
  return z;
}

// A fragment (16x32 bf16, MxK) from row-major LDS tile lds[row][stride].
__device__ inline v16bf frag_a(const __bf16* lds, int row0, int k0, int stride) {
  int lane = threadIdx.x & 31, half = lane >> 4, m = lane & 15;
  union { v16bf v; unsigned u[8]; } f;
#pragma unroll
  for (int r = 0; r < 8; ++r) {
    int kk = ((r < 4) ? (2 * r) : (16 + 2 * (r - 4))) + 8 * half;
    f.u[r] = *(const unsigned*)(lds + (row0 + m) * stride + k0 + kk);
  }
  return f.v;
}

// B fragment (32x16 bf16, KxN); LDS holds B^T row-major: lds[n*stride + k].
__device__ inline v16bf frag_bt(const __bf16* lds, int k0, int n0, int stride) {
  int lane = threadIdx.x & 31, half = lane >> 4, n = lane & 15;
  union { v16bf v; unsigned u[8]; } f;
#pragma unroll
  for (int r = 0; r < 8; ++r) {
    int kk = ((r < 4) ? (2 * r) : (16 + 2 * (r - 4))) + 8 * half;
    f.u[r] = *(const unsigned*)(lds + (n0 + n) * stride + k0 + kk);
  }
  return f.v;
}

// ---------------------------------------------------------------------------
// GEMM: out[m][n] = sum_k A[m][k] * W[n][k] + bias[n]
// Workgroup tile 256x128; 8 waves x 32 rows; K chunks of 32; software
// pipelined (next chunk prefetched into registers during WMMA).
// mode 0: A fp32 (x) -> scatter head-major Q(scaled)/K/V bf16
// mode 1: A bf16 (attn out) -> fp32 out
// ---------------------------------------------------------------------------
__global__ __launch_bounds__(256) void gemm_bf16_kernel(
    const float* __restrict__ Af, const __bf16* __restrict__ Ab,
    const float* __restrict__ W, const float* __restrict__ bias,
    int K, int N, int mode, float* __restrict__ outF,
    __bf16* __restrict__ qo, __bf16* __restrict__ ko, __bf16* __restrict__ vo)
{
  __shared__ __attribute__((aligned(16))) __bf16 sA[256 * 32];
  __shared__ __attribute__((aligned(16))) __bf16 sB[128 * 32];

  int tid = threadIdx.x;
  int wid = tid >> 5, lane = tid & 31, half = lane >> 4, nl = lane & 15;
  int mBase = blockIdx.y * 256, nBase = blockIdx.x * 128;
  int wrow = tid >> 1, wcol = (tid & 1) * 16;

  v8f acc0[8], acc1[8];
#pragma unroll
  for (int i = 0; i < 8; ++i) { acc0[i] = v8f_zero(); acc1[i] = v8f_zero(); }

  __attribute__((aligned(16))) __bf16 ra[32];   // A prefetch: one 32-wide row per thread
  __attribute__((aligned(16))) __bf16 rw[16];   // W prefetch

  auto loadTiles = [&](int k0) {
    if (mode == 0) {
      const float* src = Af + (long)(mBase + tid) * K + k0;
#pragma unroll
      for (int c = 0; c < 8; ++c) {
        float4 f = *(const float4*)(src + 4 * c);
        ra[4 * c + 0] = f2bf(f.x); ra[4 * c + 1] = f2bf(f.y);
        ra[4 * c + 2] = f2bf(f.z); ra[4 * c + 3] = f2bf(f.w);
      }
    } else {
      const __bf16* src = Ab + (long)(mBase + tid) * K + k0;
#pragma unroll
      for (int c = 0; c < 4; ++c) ((uint4*)ra)[c] = ((const uint4*)src)[c];
    }
    const float* wsrc = W + (long)(nBase + wrow) * K + k0 + wcol;
#pragma unroll
    for (int c = 0; c < 4; ++c) {
      float4 f = *(const float4*)(wsrc + 4 * c);
      rw[4 * c + 0] = f2bf(f.x); rw[4 * c + 1] = f2bf(f.y);
      rw[4 * c + 2] = f2bf(f.z); rw[4 * c + 3] = f2bf(f.w);
    }
  };

  loadTiles(0);
  int nk = K >> 5;
  for (int kt = 0; kt < nk; ++kt) {
    __syncthreads();                       // previous chunk's WMMA reads done
    {
      uint4* da = (uint4*)&sA[tid * 32];
#pragma unroll
      for (int c = 0; c < 4; ++c) da[c] = ((uint4*)ra)[c];
      uint4* db = (uint4*)&sB[wrow * 32 + wcol];
      db[0] = ((uint4*)rw)[0]; db[1] = ((uint4*)rw)[1];
    }
    __syncthreads();
    if (kt + 1 < nk) loadTiles((kt + 1) << 5);   // overlap with WMMA below

    v16bf a0 = frag_a(sA, wid * 32, 0, 32);
    v16bf a1 = frag_a(sA, wid * 32 + 16, 0, 32);
#pragma unroll
    for (int nt = 0; nt < 8; ++nt) {
      v16bf b = frag_bt(sB, 0, nt * 16, 32);
      acc0[nt] = __builtin_amdgcn_wmma_f32_16x16x32_bf16(
          false, a0, false, b, (short)0, acc0[nt], false, false);
      acc1[nt] = __builtin_amdgcn_wmma_f32_16x16x32_bf16(
          false, a1, false, b, (short)0, acc1[nt], false, false);
    }
  }

  auto epilogue = [&](v8f (&acc)[8], int mg) {
#pragma unroll
    for (int nt = 0; nt < 8; ++nt) {
      int n = nBase + nt * 16 + nl;
      float bv = bias[n];
#pragma unroll
      for (int r = 0; r < 8; ++r) {
        int m = mg + r + 8 * half;
        float val = acc[nt][r] + bv;
        if (mode == 0) {
          int sel = n >> 10, c = n & 1023, h = c >> 6, d = c & 63;
          int b_ = m >> 12, t = m & 4095;
          long dst = ((long)(b_ * N_HEAD + h) * SEQ + t) * HEAD_DIM + d;
          if (sel == 0)      qo[dst] = f2bf(val * ATT_SCALE);
          else if (sel == 1) ko[dst] = f2bf(val);
          else               vo[dst] = f2bf(val);
        } else {
          outF[(long)m * N + n] = val;
        }
      }
    }
  };
  epilogue(acc0, mBase + wid * 32);
  epilogue(acc1, mBase + wid * 32 + 16);
}

// ---------------------------------------------------------------------------
// Flash attention (causal). One workgroup per (bh, 128-row q tile); 8 waves,
// each owns 16 q rows. Double-buffered K/V tiles (K via async->LDS when
// available, V via register prefetch + transposed store). Q/P staging is
// wave-local in a shared buffer (no extra barriers).
// ---------------------------------------------------------------------------
__global__ __launch_bounds__(256) void flash_attn_kernel(
    const __bf16* __restrict__ Qg, const __bf16* __restrict__ Kg,
    const __bf16* __restrict__ Vg, __bf16* __restrict__ Og)
{
  __shared__ __attribute__((aligned(16))) __bf16 sQP[128 * 64];   // Q, then P (per-wave rows)
  __shared__ __attribute__((aligned(16))) __bf16 sK[2][64 * 64];  // [tok][d]
  __shared__ __attribute__((aligned(16))) __bf16 sV[2][64 * 64];  // [d][tok]

  int tid = threadIdx.x;
  int wid = tid >> 5, lane = tid & 31, half = lane >> 4, nl = lane & 15;
  int bh = blockIdx.y;
  int qBase = blockIdx.x * 128;

  int qrow = tid >> 1, qcol = (tid & 1) * 32;   // Q staging: wave w stages rows 16w..16w+15
  int krow = tid >> 2, kcol = (tid & 3) * 16;

  const __bf16* Qsrc  = Qg + ((long)bh * SEQ + qBase + qrow) * HEAD_DIM + qcol;
  const __bf16* Kbase = Kg + (long)bh * SEQ * HEAD_DIM;
  const __bf16* Vbase = Vg + (long)bh * SEQ * HEAD_DIM;

#if USE_ASYNC_LDS
#pragma unroll
  for (int c = 0; c < 4; ++c)
    async_ld16(&sQP[qrow * 64 + qcol + c * 8], Qsrc + c * 8);
  async_ld16(&sK[0][krow * 64 + kcol],     Kbase + krow * 64 + kcol);
  async_ld16(&sK[0][krow * 64 + kcol + 8], Kbase + krow * 64 + kcol + 8);
  __builtin_amdgcn_s_wait_asynccnt(2);   // in-order: the 4 Q loads are complete
#else
  {
    uint4* d = (uint4*)&sQP[qrow * 64 + qcol];
    const uint4* s = (const uint4*)Qsrc;
    d[0] = s[0]; d[1] = s[1]; d[2] = s[2]; d[3] = s[3];
    const __bf16* ks = Kbase + krow * 64 + kcol;
    *(uint4*)&sK[0][krow * 64 + kcol]     = *(const uint4*)ks;
    *(uint4*)&sK[0][krow * 64 + kcol + 8] = *(const uint4*)(ks + 8);
  }
#endif
  uint4 vr0, vr1;   // V register prefetch
  {
    const __bf16* vs = Vbase + krow * 64 + kcol;
    vr0 = *(const uint4*)vs; vr1 = *(const uint4*)(vs + 8);
  }

  v16bf qa0 = frag_a(sQP, wid * 16, 0, 64);    // wave-local rows
  v16bf qa1 = frag_a(sQP, wid * 16, 32, 64);

  float mrow[8], lsum[8];
#pragma unroll
  for (int r = 0; r < 8; ++r) { mrow[r] = -3.0e38f; lsum[r] = 0.0f; }
  v8f o[4];
#pragma unroll
  for (int d = 0; d < 4; ++d) o[d] = v8f_zero();

  int nj = qBase / 64 + 2;   // causal: only K blocks touching this q tile

  for (int j = 0; j < nj; ++j) {
    int buf = j & 1;
#if USE_ASYNC_LDS
    __builtin_amdgcn_s_wait_asynccnt(0);       // this tile's K loads landed
#endif
    {  // V(j): registers -> transposed LDS
      __attribute__((aligned(16))) __bf16 tmp[16];
      *(uint4*)tmp = vr0; *(uint4*)(tmp + 8) = vr1;
#pragma unroll
      for (int i = 0; i < 16; ++i) sV[buf][(kcol + i) * 64 + krow] = tmp[i];
    }
    __syncthreads();

    if (j + 1 < nj) {                          // prefetch next tile into other buffer
      long src = (long)((j + 1) * 64 + krow) * HEAD_DIM + kcol;
#if USE_ASYNC_LDS
      async_ld16(&sK[1 - buf][krow * 64 + kcol],     Kbase + src);
      async_ld16(&sK[1 - buf][krow * 64 + kcol + 8], Kbase + src + 8);
#else
      *(uint4*)&sK[1 - buf][krow * 64 + kcol]     = *(const uint4*)(Kbase + src);
      *(uint4*)&sK[1 - buf][krow * 64 + kcol + 8] = *(const uint4*)(Kbase + src + 8);
#endif
      vr0 = *(const uint4*)(Vbase + src);
      vr1 = *(const uint4*)(Vbase + src + 8);
    }

    // S = Q K^T (scale folded into Q)
    int kBase = j * 64;
    v8f s[4];
#pragma unroll
    for (int nt = 0; nt < 4; ++nt) {
      v8f st = v8f_zero();
      st = __builtin_amdgcn_wmma_f32_16x16x32_bf16(
          false, qa0, false, frag_bt(sK[buf], 0, nt * 16, 64), (short)0, st, false, false);
      st = __builtin_amdgcn_wmma_f32_16x16x32_bf16(
          false, qa1, false, frag_bt(sK[buf], 32, nt * 16, 64), (short)0, st, false, false);
      s[nt] = st;
    }

    if (kBase + 63 > qBase + wid * 16) {       // causal mask near diagonal
#pragma unroll
      for (int nt = 0; nt < 4; ++nt)
#pragma unroll
        for (int r = 0; r < 8; ++r) {
          int kp = kBase + nt * 16 + nl;
          int qp = qBase + wid * 16 + r + 8 * half;
          if (kp > qp) s[nt][r] = -3.0e38f;
        }
    }

    // online softmax: reductions across 16-lane halves
    float alpha[8], rs[8];
#pragma unroll
    for (int r = 0; r < 8; ++r) {
      float v = s[0][r];
      v = fmaxf(v, s[1][r]); v = fmaxf(v, s[2][r]); v = fmaxf(v, s[3][r]);
#pragma unroll
      for (int off = 1; off < 16; off <<= 1) v = fmaxf(v, __shfl_xor(v, off, 32));
      float mn = fmaxf(mrow[r], v);
      alpha[r] = __expf(mrow[r] - mn);
      mrow[r] = mn;
      rs[r] = 0.0f;
    }
#pragma unroll
    for (int nt = 0; nt < 4; ++nt)
#pragma unroll
      for (int r = 0; r < 8; ++r) {
        float p = __expf(s[nt][r] - mrow[r]);
        s[nt][r] = p;
        rs[r] += p;
      }
#pragma unroll
    for (int r = 0; r < 8; ++r) {
#pragma unroll
      for (int off = 1; off < 16; off <<= 1) rs[r] += __shfl_xor(rs[r], off, 32);
      lsum[r] = lsum[r] * alpha[r] + rs[r];
    }
#pragma unroll
    for (int d = 0; d < 4; ++d)
#pragma unroll
      for (int r = 0; r < 8; ++r) o[d][r] *= alpha[r];

    // P: C-layout -> A-layout via wave-local staging (LDS ops in-order per wave)
#pragma unroll
    for (int nt = 0; nt < 4; ++nt)
#pragma unroll
      for (int r = 0; r < 8; ++r)
        sQP[(wid * 16 + r + 8 * half) * 64 + nt * 16 + nl] = f2bf(s[nt][r]);

    v16bf pa0 = frag_a(sQP, wid * 16, 0, 64);
    v16bf pa1 = frag_a(sQP, wid * 16, 32, 64);
#pragma unroll
    for (int dt = 0; dt < 4; ++dt) {
      o[dt] = __builtin_amdgcn_wmma_f32_16x16x32_bf16(
          false, pa0, false, frag_bt(sV[buf], 0, dt * 16, 64), (short)0, o[dt], false, false);
      o[dt] = __builtin_amdgcn_wmma_f32_16x16x32_bf16(
          false, pa1, false, frag_bt(sV[buf], 32, dt * 16, 64), (short)0, o[dt], false, false);
    }
  }

  // normalize + store attention out as [B*T][C] bf16 for the proj GEMM
  int b_ = bh >> 4, h = bh & 15;
#pragma unroll
  for (int r = 0; r < 8; ++r) {
    float inv = 1.0f / lsum[r];
    int t = qBase + wid * 16 + r + 8 * half;
    long rowoff = ((long)b_ * SEQ + t) * N_EMBD + h * HEAD_DIM;
#pragma unroll
    for (int dt = 0; dt < 4; ++dt)
      Og[rowoff + dt * 16 + nl] = f2bf(o[dt][r] * inv);
  }
}

extern "C" void kernel_launch(void* const* d_in, const int* in_sizes, int n_in,
                              void* d_out, int out_size, void* d_ws, size_t ws_size,
                              hipStream_t stream) {
  const float* x      = (const float*)d_in[0];
  const float* qkv_w  = (const float*)d_in[1];
  const float* qkv_b  = (const float*)d_in[2];
  const float* proj_w = (const float*)d_in[3];
  const float* proj_b = (const float*)d_in[4];
  float* out = (float*)d_out;

  const long hd = (long)BHN * SEQ * HEAD_DIM;
  __bf16* q  = (__bf16*)d_ws;
  __bf16* k  = q + hd;
  __bf16* v  = k + hd;
  __bf16* ao = v + hd;   // [NTOK][N_EMBD]

  dim3 blk(256);
  gemm_bf16_kernel<<<dim3((3 * N_EMBD) / 128, NTOK / 256), blk, 0, stream>>>(
      x, nullptr, qkv_w, qkv_b, N_EMBD, 3 * N_EMBD, 0, nullptr, q, k, v);
  flash_attn_kernel<<<dim3(SEQ / 128, BHN), blk, 0, stream>>>(q, k, v, ao);
  gemm_bf16_kernel<<<dim3(N_EMBD / 128, NTOK / 256), blk, 0, stream>>>(
      nullptr, ao, proj_w, proj_b, N_EMBD, N_EMBD, 1, out, nullptr, nullptr, nullptr);
}
